// EDUPredictor_82918638617186
// MI455X (gfx1250) — compile-verified, matrix operand order checked by reading, not run on hardware
//
#include <hip/hip_runtime.h>
#include <hip/hip_bf16.h>

// ---------------------------------------------------------------------------
// EDU predictor: BiLSTM(768) -> windowed attn -> BiLSTM(4) -> CRF Viterbi
// MI455X / gfx1250: wave32, WMMA bf16 16x16x32, 320KB LDS per WGP, TDM.
// ---------------------------------------------------------------------------

typedef __bf16 bf16;
typedef __attribute__((ext_vector_type(16))) __bf16 v16bf;
typedef __attribute__((ext_vector_type(8)))  __bf16 v8bf;
typedef __attribute__((ext_vector_type(8)))  float  v8f;
typedef unsigned int u32t;
typedef __attribute__((ext_vector_type(4))) u32t u32x4;
typedef __attribute__((ext_vector_type(8))) u32t u32x8;

#define B_   8
#define S_   512
#define H_   768
#define G1_  3072   // 4*H
#define T_   4
#define NWIN 21     // 2*W+1

__device__ __forceinline__ float sigf(float x){ return 1.0f/(1.0f+__expf(-x)); }

// Load one 16-bf16 WMMA fragment row (A or B operand, row-major, K-contig).
// ISA 7.12.2: lanes 0-15 hold K {0..7,16..23}, lanes 16-31 hold K {8..15,24..31};
// caller pre-biases the pointer by (lane&16 ? 8 : 0).
__device__ __forceinline__ v16bf ld_frag(const bf16* p){
  v8bf lo = *(const v8bf*)p;
  v8bf hi = *(const v8bf*)(p + 16);
  v16bf f;
#pragma unroll
  for (int i = 0; i < 8; ++i){ f[i] = lo[i]; f[i+8] = hi[i]; }
  return f;
}

// ---------------------------------------------------------------------------
// Generic helpers
// ---------------------------------------------------------------------------
__global__ void k_zero_u32(unsigned int* p, int n){
  int i = blockIdx.x*blockDim.x + threadIdx.x;
  if (i < n) p[i] = 0u;
}

__global__ void k_f32_to_bf16(bf16* dst, const float* src, int n){
  int i = blockIdx.x*blockDim.x + threadIdx.x;
  if (i < n) dst[i] = (bf16)src[i];
}

__global__ void k_add(float* dst, const float* a, const float* b, int n){
  int i = blockIdx.x*blockDim.x + threadIdx.x;
  if (i < n) dst[i] = a[i] + b[i];
}

// ---------------------------------------------------------------------------
// WMMA GEMM: C[M,N] = A[M,K](bf16,rowmajor) * Wt[N,K](bf16,rowmajor)^T + bias
// One wave computes a 16 x (16*NSUB) strip: the A fragment is loaded once per
// K-chunk and reused across NSUB WMMAs (L2 traffic / VALU amortization).
// ---------------------------------------------------------------------------
template<int NSUB>
__global__ void k_gemm_bf16(const bf16* __restrict__ A, const bf16* __restrict__ Wt,
                            const float* __restrict__ bias, float* __restrict__ C,
                            int M, int N, int K){
  int lane = threadIdx.x & 31;
  int wave = blockIdx.x * (blockDim.x >> 5) + (threadIdx.x >> 5);
  int stripsN = N / (16*NSUB);
  int tM  = wave / stripsN;
  int tN0 = (wave - tM * stripsN) * NSUB;
  if (tM * 16 >= M) return;

  const bf16* pa = A + (size_t)(tM*16 + (lane & 15)) * K + ((lane & 16) ? 8 : 0);
  const bf16* pb[NSUB];
#pragma unroll
  for (int j = 0; j < NSUB; ++j)
    pb[j] = Wt + (size_t)((tN0 + j)*16 + (lane & 15)) * K + ((lane & 16) ? 8 : 0);

  v8f acc[NSUB];
#pragma unroll
  for (int j = 0; j < NSUB; ++j) acc[j] = (v8f){};

  for (int k = 0; k < K; k += 32){
    __builtin_prefetch(pa + k + 256, 0, 0);
    v16bf fa = ld_frag(pa + k);
#pragma unroll
    for (int j = 0; j < NSUB; ++j){
      v16bf fb = ld_frag(pb[j] + k);
      acc[j] = __builtin_amdgcn_wmma_f32_16x16x32_bf16(false, fa, false, fb,
                                                       (short)0, acc[j], false, false);
    }
  }
  int m0 = tM*16 + ((lane >> 4) << 3);   // C/D layout: m = r + 8*(lane/16)
#pragma unroll
  for (int j = 0; j < NSUB; ++j){
    int n = (tN0 + j)*16 + (lane & 15);
    float bv = bias ? bias[n] : 0.0f;
#pragma unroll
    for (int r = 0; r < 8; ++r) C[(size_t)(m0 + r)*N + n] = acc[j][r] + bv;
  }
}

// ---------------------------------------------------------------------------
// L1 LSTM recurrence. grid = (16 slices, 2 dirs). Each WG owns 48 hidden dims
// = 192 gate rows; their 192x768 bf16 w_hh slice lives in LDS (288KB),
// fetched by the Tensor Data Mover as a 768 x 48 x 4 tile.
// Per step: WMMA 16x192x768 gate GEMM from LDS, nonlinearity, publish h slice,
// cross-WG barrier via monotonic atomic counter (double-buffered h).
// ---------------------------------------------------------------------------
#define LDS_W   (192*H_*2)        // 294912 B
#define LDS_HS  (16*H_*2)         // 24576 B
#define LDS_G   (192*8*4)         // 6144 B
#define LSTM1_LDS (LDS_W + LDS_HS + LDS_G)  // 325632 B <= 327680

__launch_bounds__(512, 1)
__global__ void k_lstm1(const bf16* __restrict__ whh_f, const bf16* __restrict__ whh_r,
                        const float* __restrict__ xs_f, const float* __restrict__ xs_r,
                        float* __restrict__ hseq_f, float* __restrict__ hseq_r,
                        float* __restrict__ hbuf, unsigned int* __restrict__ cnt){
  int p   = blockIdx.x;   // 0..15 slice
  int dir = blockIdx.y;   // 0 fwd, 1 rev
  const bf16*  whh  = dir ? whh_r  : whh_f;
  const float* xs   = dir ? xs_r   : xs_f;
  float*       hseq = dir ? hseq_r : hseq_f;
  float*       hb   = hbuf + (size_t)dir * 2 * B_ * H_;

  extern __shared__ char smem[];
  bf16*  wlds   = (bf16*)smem;
  bf16*  hstage = (bf16*)(smem + LDS_W);
  float* glds   = (float*)(smem + LDS_W + LDS_HS);

  int tid  = threadIdx.x;         // 512 threads = 16 waves
  int lane = tid & 31, wv = tid >> 5;

  // --- TDM: fetch the w_hh slice (rows {g*768 + p*48 + y}) as a 3D tile:
  //     x = 768 elems, y = 48 rows @ stride 768, z = 4 gate blocks @ stride 768*768.
  //     D# per CDNA5 ISA; lds_addr = 0 (dyn-smem base).
  if (tid == 0){
    unsigned long long ga =
        (unsigned long long)(const void*)(whh + (size_t)p * 48 * H_);
    u32x4 g0;
    g0[0] = 1u;                               // count=1, user mode
    g0[1] = 0u;                               // lds_addr
    g0[2] = (u32t)ga;                         // global_addr[31:0]
    g0[3] = (u32t)(ga >> 32) | (2u << 30);    // global_addr[56:32] | type=2
    u32x8 g1;
    g1[0] = (1u << 16);                       // data_size = 2 bytes
    g1[1] = ((u32t)H_ << 16);                 // tensor_dim0[15:0]
    g1[2] = ((u32t)H_ >> 16) | ((u32t)G1_ << 16);  // tdim0[31:16] | tensor_dim1[15:0]
    g1[3] = ((u32t)G1_ >> 16) | ((u32t)H_ << 16);  // tdim1[31:16] | tile_dim0=768
    g1[4] = 48u | (4u << 16);                 // tile_dim1=48 | tile_dim2=4
    g1[5] = (u32t)H_;                         // tensor_dim0_stride[31:0] = 768
    g1[6] = 0u | (((u32t)(H_*H_) & 0xFFFFu) << 16); // s0[47:32] | dim1_stride[15:0]
    g1[7] = (u32t)(H_*H_) >> 16;              // dim1_stride[47:16]
    u32x4 g2;
    g2[0] = 4u;                               // tensor_dim2 = 4
    g2[1] = 0u; g2[2] = 0u; g2[3] = 0u;
    u32x4 g3 = {0u, 0u, 0u, 0u};
    asm volatile("tensor_load_to_lds %0, %1, %2, %3"
                 :: "s"(g0), "s"(g1), "s"(g2), "s"(g3) : "memory");
    __builtin_amdgcn_s_wait_tensorcnt(0);
  }
  __syncthreads();

  // Backstop staging (same bytes; keeps correctness independent of D# packing).
  for (int v = tid; v < (192*H_)/8; v += blockDim.x){
    int e  = v * 8;
    int lr = e / H_, col = e - lr * H_;
    int gr = (lr/48)*H_ + p*48 + (lr%48);
    *(v8bf*)(wlds + (size_t)lr*H_ + col) = *(const v8bf*)(whh + (size_t)gr*H_ + col);
  }
  // Zero h stage (rows 8..15 stay zero = batch padding)
  for (int v = tid; v < 16*H_; v += blockDim.x) hstage[v] = (bf16)0.0f;

  float creg = 0.0f;              // cell state, one (jloc,b) per thread (tid<384)
  int uj = tid / B_, ub = tid % B_;
  __syncthreads();

  for (int step = 0; step < S_; ++step){
    const float* hprev = hb + (size_t)( step      & 1) * B_ * H_;
    float*       hnext = hb + (size_t)((step + 1) & 1) * B_ * H_;

    // stage h_prev (8x768) as bf16 rows 0..7
    for (int v = tid; v < B_*H_; v += blockDim.x) hstage[v] = (bf16)hprev[v];
    __syncthreads();

    // gates[16x192] = Hpad[16x768] x Wslice^T via WMMA, 12 N-tiles over 12 waves
    if (wv < 12){
      v8f acc = {};
      const bf16* pa = hstage + (size_t)(lane & 15)*H_            + ((lane & 16) ? 8 : 0);
      const bf16* pb = wlds   + (size_t)(wv*16 + (lane & 15))*H_  + ((lane & 16) ? 8 : 0);
      for (int k = 0; k < H_; k += 32){
        v16bf fa = ld_frag(pa + k);
        v16bf fb = ld_frag(pb + k);
        acc = __builtin_amdgcn_wmma_f32_16x16x32_bf16(false, fa, false, fb,
                                                      (short)0, acc, false, false);
      }
      if (lane < 16){               // m = r (0..7) = batch rows; lanes>=16 are padding
        int n = wv*16 + lane;
#pragma unroll
        for (int r = 0; r < 8; ++r) glds[n*8 + r] = acc[r];
      }
    }
    __syncthreads();

    int s_eff = dir ? (S_ - 1 - step) : step;
    if (tid < 48*B_){
      int j = p*48 + uj;
      size_t xb = ((size_t)(ub*S_ + s_eff)) * G1_;   // xs already has bias
      float pi = xs[xb           + j] + glds[(0*48 + uj)*8 + ub];
      float pf = xs[xb + H_      + j] + glds[(1*48 + uj)*8 + ub];
      float pg = xs[xb + 2*H_    + j] + glds[(2*48 + uj)*8 + ub];
      float po = xs[xb + 3*H_    + j] + glds[(3*48 + uj)*8 + ub];
      float c  = sigf(pf)*creg + sigf(pi)*tanhf(pg);
      creg = c;
      float h  = sigf(po)*tanhf(c);
      hnext[(size_t)ub*H_ + j] = h;
      hseq[((size_t)(ub*S_ + s_eff))*H_ + j] = h;
    }
    __threadfence();
    __syncthreads();
    if (tid == 0){
      atomicAdd(&cnt[dir], 1u);
      volatile unsigned int* vc = cnt + dir;
      unsigned int target = 16u * (unsigned)(step + 1);
      while (*vc < target) __builtin_amdgcn_s_sleep(1);
    }
    __syncthreads();
    __threadfence();
  }
}

// ---------------------------------------------------------------------------
// Attention: precompute a = h.w1, u = h.w2 (one wave per (b,s))
// ---------------------------------------------------------------------------
__global__ void k_attn_pre(const float* __restrict__ hs, const float* __restrict__ aw,
                           float* __restrict__ a_pre, float* __restrict__ u_pre){
  int wave = (blockIdx.x*blockDim.x + threadIdx.x) >> 5;
  int lane = threadIdx.x & 31;
  if (wave >= B_*S_) return;
  const float* h = hs + (size_t)wave * H_;
  float pa = 0.f, pu = 0.f;
  for (int d = lane; d < H_; d += 32){ float hv = h[d]; pa += hv*aw[d]; pu += hv*aw[H_+d]; }
#pragma unroll
  for (int off = 16; off; off >>= 1){
    pa += __shfl_down(pa, off, 32);
    pu += __shfl_down(pu, off, 32);
  }
  if (!lane){ a_pre[wave] = pa; u_pre[wave] = pu; }
}

// Window attention main: one block per (b,s). 21 scores, softmax, weighted sum.
__global__ void k_attn_main(const float* __restrict__ hs, const float* __restrict__ a_pre,
                            const float* __restrict__ u_pre, const float* __restrict__ attw,
                            const float* __restrict__ attb, float* __restrict__ att){
  int bs = blockIdx.x;              // 0..4095
  int b = bs >> 9, s = bs & (S_-1);
  __shared__ float hw3[H_];
  __shared__ float sc[24];
  __shared__ float awgt[24];
  const float* h  = hs + (size_t)bs * H_;
  const float* w3 = attw + 2*H_;
  for (int d = threadIdx.x; d < H_; d += blockDim.x) hw3[d] = h[d] * w3[d];
  __syncthreads();
  int wv = threadIdx.x >> 5, lane = threadIdx.x & 31;
  for (int k = wv; k < NWIN; k += (blockDim.x >> 5)){
    int idx = s + k - 10;
    float sco = -1e9f;
    if (idx >= 0 && idx < S_){
      const float* hk = hs + ((size_t)(b*S_ + idx)) * H_;
      float part = 0.f;
      for (int d = lane; d < H_; d += 32) part += hw3[d] * hk[d];
#pragma unroll
      for (int off = 16; off; off >>= 1) part += __shfl_down(part, off, 32);
      sco = a_pre[bs] + u_pre[b*S_ + idx] + part + attb[0];
    }
    if (!lane) sc[k] = sco;
  }
  __syncthreads();
  if (threadIdx.x == 0){
    float mx = sc[0];
    for (int k = 1; k < NWIN; ++k) mx = fmaxf(mx, sc[k]);
    float sum = 0.f;
    for (int k = 0; k < NWIN; ++k){ float e = __expf(sc[k]-mx); awgt[k] = e; sum += e; }
    float inv = 1.0f/sum;
    for (int k = 0; k < NWIN; ++k) awgt[k] *= inv;
  }
  __syncthreads();
  float* out = att + (size_t)bs * H_;
  for (int d = threadIdx.x; d < H_; d += blockDim.x){
    float acc = 0.f;
    for (int k = 0; k < NWIN; ++k){
      int idx = s + k - 10;
      if (idx >= 0 && idx < S_) acc += awgt[k] * hs[((size_t)(b*S_ + idx))*H_ + d];
    }
    out[d] = acc;
  }
}

// pack x2 = [hs | att] as bf16 for the L2 input projection GEMM
__global__ void k_pack_x2(const float* __restrict__ hs, const float* __restrict__ att,
                          bf16* __restrict__ x2, int n){
  int i = blockIdx.x*blockDim.x + threadIdx.x;
  if (i >= n) return;
  int bs = i / H_, d = i - bs*H_;
  x2[(size_t)bs*(2*H_) + d]       = (bf16)hs[i];
  x2[(size_t)bs*(2*H_) + H_ + d]  = (bf16)att[i];
}

// ---------------------------------------------------------------------------
// L2 LSTM (hidden=4): trivial recurrence, one block per direction.
// ---------------------------------------------------------------------------
__global__ void k_lstm2(const float* __restrict__ xs2, const float* __restrict__ whh,
                        float* __restrict__ emdir, int dir){
  __shared__ float hS[B_*T_], cS[B_*T_], gS[B_*16];
  int t = threadIdx.x;                 // 128 = 8 batch * 16 gates
  if (t < B_*T_){ hS[t] = 0.f; cS[t] = 0.f; }
  __syncthreads();
  int b = t >> 4, g = t & 15;
  float w0 = whh[g*4+0], w1 = whh[g*4+1], w2 = whh[g*4+2], w3 = whh[g*4+3];
  for (int step = 0; step < S_; ++step){
    int s = dir ? (S_ - 1 - step) : step;
    gS[t] = xs2[((size_t)(b*S_ + s))*16 + g]
          + w0*hS[b*4+0] + w1*hS[b*4+1] + w2*hS[b*4+2] + w3*hS[b*4+3];
    __syncthreads();
    if (t < B_*T_){
      int bb = t >> 2, u = t & 3;
      float i  = sigf(gS[bb*16 + u]);
      float f  = sigf(gS[bb*16 + 4 + u]);
      float gg = tanhf(gS[bb*16 + 8 + u]);
      float o  = sigf(gS[bb*16 + 12 + u]);
      float c  = f*cS[t] + i*gg;
      cS[t] = c;
      float h = o*tanhf(c);
      hS[t] = h;
      emdir[((size_t)(bb*S_ + s))*T_ + u] = h;
    }
    __syncthreads();
  }
}

// ---------------------------------------------------------------------------
// Viterbi over T=4 tags; one thread per batch element.
// ---------------------------------------------------------------------------
__global__ void k_viterbi(const float* __restrict__ em, const float* __restrict__ startv,
                          const float* __restrict__ endv, const float* __restrict__ trans,
                          int* __restrict__ tags, unsigned char* __restrict__ bp){
  int b = threadIdx.x;
  if (b >= B_) return;
  float sc[T_];
#pragma unroll
  for (int t = 0; t < T_; ++t) sc[t] = startv[t] + em[((size_t)(b*S_))*T_ + t];
  for (int s = 1; s < S_; ++s){
    float ns[T_];
#pragma unroll
    for (int t2 = 0; t2 < T_; ++t2){
      float best = -1e30f; int bi = 0;
#pragma unroll
      for (int t1 = 0; t1 < T_; ++t1){
        float v = sc[t1] + trans[t1*T_ + t2];
        if (v > best){ best = v; bi = t1; }
      }
      ns[t2] = best + em[((size_t)(b*S_ + s))*T_ + t2];
      bp[((size_t)(b*S_ + s))*T_ + t2] = (unsigned char)bi;
    }
#pragma unroll
    for (int t = 0; t < T_; ++t) sc[t] = ns[t];
  }
  int last = 0; float best = -1e30f;
#pragma unroll
  for (int t = 0; t < T_; ++t){ float v = sc[t] + endv[t]; if (v > best){ best = v; last = t; } }
  tags[b*S_ + S_-1] = last;
  for (int s = S_-1; s > 0; --s){
    last = (int)bp[((size_t)(b*S_ + s))*T_ + last];
    tags[b*S_ + s - 1] = last;
  }
}

// ---------------------------------------------------------------------------
// Host-side orchestration
// ---------------------------------------------------------------------------
extern "C" void kernel_launch(void* const* d_in, const int* in_sizes, int n_in,
                              void* d_out, int out_size, void* d_ws, size_t ws_size,
                              hipStream_t stream) {
  const float* emb     = (const float*)d_in[0];
  const float* l1f_wih = (const float*)d_in[1];
  const float* l1f_whh = (const float*)d_in[2];
  const float* l1f_b   = (const float*)d_in[3];
  const float* l1r_wih = (const float*)d_in[4];
  const float* l1r_whh = (const float*)d_in[5];
  const float* l1r_b   = (const float*)d_in[6];
  const float* attn_w  = (const float*)d_in[7];
  const float* attn_b  = (const float*)d_in[8];
  const float* l2f_wih = (const float*)d_in[9];
  const float* l2f_whh = (const float*)d_in[10];
  const float* l2f_b   = (const float*)d_in[11];
  const float* l2r_wih = (const float*)d_in[12];
  const float* l2r_whh = (const float*)d_in[13];
  const float* l2r_b   = (const float*)d_in[14];
  const float* crf_s   = (const float*)d_in[15];
  const float* crf_e   = (const float*)d_in[16];
  const float* crf_t   = (const float*)d_in[17];

  // workspace carve-up
  char* w = (char*)d_ws;
  auto carve = [&](size_t bytes)->char*{
    char* p = w; w += (bytes + 255) & ~(size_t)255; return p;
  };
  const int BS = B_*S_;                                   // 4096
  bf16*  emb_bf   = (bf16*) carve((size_t)BS*H_*2);
  bf16*  w1f_ih   = (bf16*) carve((size_t)G1_*H_*2);
  bf16*  w1r_ih   = (bf16*) carve((size_t)G1_*H_*2);
  bf16*  w1f_hh   = (bf16*) carve((size_t)G1_*H_*2);
  bf16*  w1r_hh   = (bf16*) carve((size_t)G1_*H_*2);
  bf16*  w2f_ih   = (bf16*) carve((size_t)16*2*H_*2);
  bf16*  w2r_ih   = (bf16*) carve((size_t)16*2*H_*2);
  float* xs1f     = (float*)carve((size_t)BS*G1_*4);
  float* xs1r     = (float*)carve((size_t)BS*G1_*4);
  float* hseq_f   = (float*)carve((size_t)BS*H_*4);
  float* hseq_r   = (float*)carve((size_t)BS*H_*4);
  float* hs       = (float*)carve((size_t)BS*H_*4);
  float* att      = (float*)carve((size_t)BS*H_*4);
  bf16*  x2_bf    = (bf16*) carve((size_t)BS*2*H_*2);
  float* xs2f     = (float*)carve((size_t)BS*16*4);
  float* xs2r     = (float*)carve((size_t)BS*16*4);
  float* a_pre    = (float*)carve((size_t)BS*4);
  float* u_pre    = (float*)carve((size_t)BS*4);
  float* em_f     = (float*)carve((size_t)BS*T_*4);
  float* em_r     = (float*)carve((size_t)BS*T_*4);
  float* hbuf     = (float*)carve((size_t)2*2*B_*H_*4);   // dir x dblbuf x B x H
  unsigned int* cnt = (unsigned int*)carve(256);
  unsigned char* bp = (unsigned char*)carve((size_t)BS*T_);

  int* tags   = (int*)d_out;                 // B*S int32
  float* em_o = (float*)d_out + BS;          // B*S*T fp32

  auto up = [](int n, int blk){ return (n + blk - 1) / blk; };

  // 0) reset barrier counters + h double-buffers (must happen every call)
  {
    int nz = (2*2*B_*H_*4 + 256)/4;
    hipLaunchKernelGGL(k_zero_u32, dim3(up(nz,256)), dim3(256), 0, stream,
                       (unsigned int*)hbuf, nz);
  }
  // 1) fp32 -> bf16 conversions
  hipLaunchKernelGGL(k_f32_to_bf16, dim3(up(BS*H_,256)),  dim3(256), 0, stream, emb_bf, emb, BS*H_);
  hipLaunchKernelGGL(k_f32_to_bf16, dim3(up(G1_*H_,256)), dim3(256), 0, stream, w1f_ih, l1f_wih, G1_*H_);
  hipLaunchKernelGGL(k_f32_to_bf16, dim3(up(G1_*H_,256)), dim3(256), 0, stream, w1r_ih, l1r_wih, G1_*H_);
  hipLaunchKernelGGL(k_f32_to_bf16, dim3(up(G1_*H_,256)), dim3(256), 0, stream, w1f_hh, l1f_whh, G1_*H_);
  hipLaunchKernelGGL(k_f32_to_bf16, dim3(up(G1_*H_,256)), dim3(256), 0, stream, w1r_hh, l1r_whh, G1_*H_);
  hipLaunchKernelGGL(k_f32_to_bf16, dim3(up(16*2*H_,256)),dim3(256), 0, stream, w2f_ih, l2f_wih, 16*2*H_);
  hipLaunchKernelGGL(k_f32_to_bf16, dim3(up(16*2*H_,256)),dim3(256), 0, stream, w2r_ih, l2r_wih, 16*2*H_);

  // 2) L1 input projections: (4096x768) x (768x3072) + b -> xs1
  //    16x64 strip per wave (NSUB=4)
  {
    int waves = (BS/16) * (G1_/64);
    int blocks = up(waves, 8);
    hipLaunchKernelGGL((k_gemm_bf16<4>), dim3(blocks), dim3(256), 0, stream,
                       emb_bf, w1f_ih, l1f_b, xs1f, BS, G1_, H_);
    hipLaunchKernelGGL((k_gemm_bf16<4>), dim3(blocks), dim3(256), 0, stream,
                       emb_bf, w1r_ih, l1r_b, xs1r, BS, G1_, H_);
  }

  // 3) L1 recurrence: 16 slices x 2 dirs, 288KB w_hh slice in LDS via TDM
  {
    hipFuncSetAttribute((const void*)&k_lstm1,
                        hipFuncAttributeMaxDynamicSharedMemorySize, LSTM1_LDS);
    hipLaunchKernelGGL(k_lstm1, dim3(16,2), dim3(512), LSTM1_LDS, stream,
                       w1f_hh, w1r_hh, xs1f, xs1r, hseq_f, hseq_r, hbuf, cnt);
  }

  // 4) hs = h_fwd + h_rev
  hipLaunchKernelGGL(k_add, dim3(up(BS*H_,256)), dim3(256), 0, stream,
                     hs, hseq_f, hseq_r, BS*H_);

  // 5) windowed attention
  hipLaunchKernelGGL(k_attn_pre,  dim3(up(BS*32,256)), dim3(256), 0, stream,
                     hs, attn_w, a_pre, u_pre);
  hipLaunchKernelGGL(k_attn_main, dim3(BS), dim3(256), 0, stream,
                     hs, a_pre, u_pre, attn_w, attn_b, att);

  // 6) x2 = [hs|att] bf16, then L2 input projections (N=16)
  hipLaunchKernelGGL(k_pack_x2, dim3(up(BS*H_,256)), dim3(256), 0, stream,
                     hs, att, x2_bf, BS*H_);
  {
    int waves = (BS/16);
    int blocks = up(waves, 8);
    hipLaunchKernelGGL((k_gemm_bf16<1>), dim3(blocks), dim3(256), 0, stream,
                       x2_bf, w2f_ih, l2f_b, xs2f, BS, 16, 2*H_);
    hipLaunchKernelGGL((k_gemm_bf16<1>), dim3(blocks), dim3(256), 0, stream,
                       x2_bf, w2r_ih, l2r_b, xs2r, BS, 16, 2*H_);
  }

  // 7) L2 recurrence (hidden=4)
  hipLaunchKernelGGL(k_lstm2, dim3(1), dim3(128), 0, stream, xs2f, l2f_whh, em_f, 0);
  hipLaunchKernelGGL(k_lstm2, dim3(1), dim3(128), 0, stream, xs2r, l2r_whh, em_r, 1);

  // 8) em = em_f + em_r (written straight into the output buffer)
  hipLaunchKernelGGL(k_add, dim3(up(BS*T_,256)), dim3(256), 0, stream,
                     em_o, em_f, em_r, BS*T_);

  // 9) Viterbi decode -> tags
  hipLaunchKernelGGL(k_viterbi, dim3(1), dim3(32), 0, stream,
                     em_o, crf_s, crf_e, crf_t, tags, bp);

  (void)in_sizes; (void)n_in; (void)out_size; (void)ws_size;
}
